// FastQuantShareQK_64931315581619
// MI455X (gfx1250) — matrix operand order repaired and checked by reading, not compiled, not run on hardware
//
#include <hip/hip_runtime.h>
#include <math.h>

// FastQuantShareQK (Performer feature-map + global-threshold binarization).
// 3 passes (global max -> global exp-sum -> binarize), each recomputing the
// thin GEMM X[T,64] x P^T[64,256] with f16 WMMA using a hi/lo split
// (Ah*Bh + Al*Bh + Ah*Bl) for fp32-class accuracy at matrix-core rates.

typedef _Float16 f16;
typedef __attribute__((ext_vector_type(16))) _Float16 v16h;
typedef __attribute__((ext_vector_type(8))) float v8f;

#define DIMH 64
#define NF 256
#define BLOCK_THREADS 256
#define WAVES_PER_BLOCK 8
#define TPW 2  // 16-token tiles processed per wave per B-fragment load

__device__ __forceinline__ unsigned int f32_sortable(float f) {
    unsigned int u = __float_as_uint(f);
    return (u & 0x80000000u) ? ~u : (u | 0x80000000u);
}
__device__ __forceinline__ float sortable_f32(unsigned int u) {
    return (u & 0x80000000u) ? __uint_as_float(u & 0x7FFFFFFFu)
                             : __uint_as_float(~u);
}

__global__ void init_ws_kernel(unsigned int* g_maxbits, double* g_sum) {
    *g_maxbits = 0u;  // sortable-min (below every real float)
    *g_sum = 0.0;
}

#define WMMA_F16(A, B, C)                                            \
    __builtin_amdgcn_wmma_f32_16x16x32_f16(false, (A), false, (B),   \
                                           (short)0, (C), false, false)

// MODE 0: global max of data_dash
// MODE 1: global sum of exp(data_dash - diag - M)
// MODE 2: write binarized output
template <int MODE>
__global__ void __launch_bounds__(BLOCK_THREADS)
performer_pass(const float* __restrict__ qk,
               const float* __restrict__ proj,
               float* __restrict__ out,
               unsigned int* __restrict__ g_maxbits,
               double* __restrict__ g_sum,
               int ntiles) {
    // Projection staged as f16 hi/lo planes: [j][d] row-major, 32KB each.
    __shared__ __align__(64) f16 ldsProjH[NF * DIMH];
    __shared__ __align__(64) f16 ldsProjL[NF * DIMH];
    __shared__ double blkSum;

    for (int i = threadIdx.x; i < NF * DIMH; i += BLOCK_THREADS) {
        float p = proj[i];
        f16 h = (f16)p;
        ldsProjH[i] = h;
        ldsProjL[i] = (f16)(p - (float)h);
    }
    if (MODE == 1 && threadIdx.x == 0) blkSum = 0.0;
    __syncthreads();

    const int lane = threadIdx.x & 31;
    const int half = lane >> 4;  // which 16-lane half of the wave
    const int mrow = lane & 15;  // A row / B,C column index for this lane
    const float dn = 0.35355339059327379f;  // 64^-0.25

    float Mglob = 0.0f, meanE = 0.0f;
    if (MODE >= 1) Mglob = sortable_f32(*g_maxbits);
    if (MODE == 2)
        meanE = (float)(*g_sum * (1.0 / ((double)ntiles * 16.0 * (double)NF)));

    float waveMax = -3.4e38f;
    float laneSum = 0.0f;

    const int waveGlobal = blockIdx.x * WAVES_PER_BLOCK + (threadIdx.x >> 5);
    const int waveCount = gridDim.x * WAVES_PER_BLOCK;
    const int groups = ntiles / TPW;

    for (int g = waveGlobal; g < groups; g += waveCount) {
        const int tokbase = g * (16 * TPW);

        // ---- Load + split A for TPW tiles.
        // 16-bit A 16x32 layout (matches B): lane element e <-> K = 32c
        // + 16*half + e: one contiguous 16-float run per chunk.
        v16h Ah[TPW][2], Al[TPW][2];
        float dOut[TPW][8];
#pragma unroll
        for (int t = 0; t < TPW; ++t) {
            const float* qr =
                qk + ((size_t)(tokbase + 16 * t + mrow)) * DIMH;
            float sA = 0.0f;
#pragma unroll
            for (int c = 0; c < 2; ++c) {
#pragma unroll
                for (int e = 0; e < 16; ++e) {
                    int k = 32 * c + 16 * half + e;
                    float x = qr[k] * dn;
                    f16 xh = (f16)x;
                    Ah[t][c][e] = xh;
                    Al[t][c][e] = (f16)(x - (float)xh);
                    sA = __builtin_fmaf(x, x, sA);
                }
            }
            // diag = 0.5*||dn*q||^2; each half-wave saw half the K range.
            if (MODE >= 1) {
                sA += __shfl_xor(sA, 16, 32);
                float diagRow = 0.5f * sA;  // diag of token tokbase+16t+mrow
#pragma unroll
                for (int v = 0; v < 8; ++v)
                    dOut[t][v] = __shfl(diagRow, v + 8 * half, 32);
            }
        }

#pragma unroll 1
        for (int nt = 0; nt < 16; ++nt) {
            // 16-bit B 32x16 layout used here (matches A): element e <-> K = 32c + 16*half + e,
            // one contiguous 32B run per chunk per plane.
            const v16h* bh = (const v16h*)(ldsProjH +
                                           (nt * 16 + mrow) * DIMH + 16 * half);
            const v16h* bl = (const v16h*)(ldsProjL +
                                           (nt * 16 + mrow) * DIMH + 16 * half);
            v16h Bh0 = bh[0], Bh1 = bh[2];
            v16h Bl0 = bl[0], Bl1 = bl[2];

#pragma unroll
            for (int t = 0; t < TPW; ++t) {
                v8f c = {};
                // Split-product GEMM: Ah*Bh + Al*Bh + Ah*Bl (~fp32 accurate)
                c = WMMA_F16(Ah[t][0], Bh0, c);
                c = WMMA_F16(Al[t][0], Bh0, c);
                c = WMMA_F16(Ah[t][0], Bl0, c);
                c = WMMA_F16(Ah[t][1], Bh1, c);
                c = WMMA_F16(Al[t][1], Bh1, c);
                c = WMMA_F16(Ah[t][1], Bl1, c);

                if (MODE == 0) {
#pragma unroll
                    for (int v = 0; v < 8; ++v)
                        waveMax = fmaxf(waveMax, c[v]);
                } else if (MODE == 1) {
#pragma unroll
                    for (int v = 0; v < 8; ++v)
                        laneSum += __expf(c[v] - dOut[t][v] - Mglob);
                } else {
#pragma unroll
                    for (int v = 0; v < 8; ++v) {
                        float e = __expf(c[v] - dOut[t][v] - Mglob);
                        int m = v + 8 * half;  // C layout: VGPR v, half -> row
                        out[((size_t)(tokbase + 16 * t + m)) * NF + nt * 16 +
                            mrow] = (e > meanE) ? 1.0f : 0.0f;
                    }
                }
            }
        }
    }

    if (MODE == 0) {
#pragma unroll
        for (int off = 16; off >= 1; off >>= 1)
            waveMax = fmaxf(waveMax, __shfl_xor(waveMax, off, 32));
        if (lane == 0) atomicMax(g_maxbits, f32_sortable(waveMax));
    } else if (MODE == 1) {
#pragma unroll
        for (int off = 16; off >= 1; off >>= 1)
            laneSum += __shfl_xor(laneSum, off, 32);
        if (lane == 0) atomicAdd(&blkSum, (double)laneSum);
        __syncthreads();
        if (threadIdx.x == 0) atomicAdd(g_sum, blkSum);
    }
}

extern "C" void kernel_launch(void* const* d_in, const int* in_sizes, int n_in,
                              void* d_out, int out_size, void* d_ws,
                              size_t ws_size, hipStream_t stream) {
    const float* qk = (const float*)d_in[0];    // [4,8,8192,64] fp32
    const float* proj = (const float*)d_in[1];  // [256,64] fp32
    float* out = (float*)d_out;                 // [4,8,8192,256] fp32 (0/1)

    unsigned int* g_maxbits = (unsigned int*)d_ws;
    double* g_sum = (double*)((char*)d_ws + 8);

    const int tokens = in_sizes[0] / DIMH;  // 262144
    const int ntiles = tokens / 16;         // 16384

    init_ws_kernel<<<1, 1, 0, stream>>>(g_maxbits, g_sum);

    int blocks = 1024;
    int groups = ntiles / TPW;
    int needed = (groups + WAVES_PER_BLOCK - 1) / WAVES_PER_BLOCK;
    if (blocks > needed) blocks = needed;

    performer_pass<0><<<blocks, BLOCK_THREADS, 0, stream>>>(
        qk, proj, out, g_maxbits, g_sum, ntiles);
    performer_pass<1><<<blocks, BLOCK_THREADS, 0, stream>>>(
        qk, proj, out, g_maxbits, g_sum, ntiles);
    performer_pass<2><<<blocks, BLOCK_THREADS, 0, stream>>>(
        qk, proj, out, g_maxbits, g_sum, ntiles);
}
